// Model_82935818486476
// MI455X (gfx1250) — compile-verified, matrix operand order checked by reading, not run on hardware
//
#include <hip/hip_runtime.h>
#include <hip/hip_bf16.h>

// ---------- types ----------
typedef __attribute__((ext_vector_type(16))) __bf16 v16bf;
typedef __attribute__((ext_vector_type(8)))  __bf16 v8bf;
typedef __attribute__((ext_vector_type(8)))  float  v8f;

// Problem dims
#define NB 64      // batch
#define NS 512     // seq
#define NE 768     // embed
#define NH 256     // hidden
#define NG 1024    // 4*H
#define NT 10      // tags
#define NTP 16     // padded tags

// ---------- WMMA helpers ----------
__device__ __forceinline__ v8f wmma_bf16(v16bf a, v16bf b, v8f c) {
  return __builtin_amdgcn_wmma_f32_16x16x32_bf16(false, a, false, b, (short)0, c, false, false);
}

// A/B operand (16x32 bf16 tile, row-major source of shape [rows x ld]).
// Lane L (0-15): M/N = L, K-half 0 ; lane 16-31: M/N = L-16, K-half 8.
// VGPR i<4 -> K = 2i+kh, {+0,+1}; VGPR i>=4 -> K = 16+2(i-4)+kh.
__device__ __forceinline__ v16bf load_tile_bf16(const __bf16* base, int ld,
                                                int r0, int k0, int lane) {
  int m  = lane & 15;
  int kh = (lane >> 4) << 3;
  const __bf16* p = base + (size_t)(r0 + m) * ld + (k0 + kh);
  v8bf lo = *reinterpret_cast<const v8bf*>(p);
  v8bf hi = *reinterpret_cast<const v8bf*>(p + 16);
  v16bf out;
#pragma unroll
  for (int i = 0; i < 8; ++i) { out[i] = lo[i]; out[8 + i] = hi[i]; }
  return out;
}

__device__ __forceinline__ float sigm(float x) { return 1.f / (1.f + __expf(-x)); }

// ---------- kernel 0: fp32 -> bf16 conversion (weights and activations) ----------
__global__ void cvt_f32_to_bf16(const float* __restrict__ src,
                                __bf16* __restrict__ dst, int n) {
  int i = blockIdx.x * blockDim.x + threadIdx.x;
  if (i < n) dst[i] = (__bf16)src[i];
}

// lin_w (10 x 512) -> zero-padded bf16 (16 x 512)
__global__ void pad_linw(const float* __restrict__ src, __bf16* __restrict__ dst) {
  int i = blockIdx.x * blockDim.x + threadIdx.x;
  if (i >= NTP * 2 * NH) return;
  int row = i >> 9, col = i & 511;
  float v = (row < NT) ? src[row * (2 * NH) + col] : 0.f;
  dst[i] = (__bf16)v;
}

// ---------- kernel 1: xg = x @ w_ih^T + b_ih + b_hh (both dirs), bf16 out ----------
// Wave computes a 16(M) x 128(N) strip: 1 A-tile reused across 8 N-tiles.
// A: x rows r = b*NS + s (ld=NE).  Output row = s*NB + b (layout [s][b][g], bf16).
__global__ void xg_gemm(const __bf16* __restrict__ xbf,
                        const __bf16* __restrict__ w_f, const __bf16* __restrict__ w_b,
                        const float* __restrict__ bih_f, const float* __restrict__ bhh_f,
                        const float* __restrict__ bih_b, const float* __restrict__ bhh_b,
                        __bf16* __restrict__ xg_f, __bf16* __restrict__ xg_b) {
  int wave = blockIdx.x * (blockDim.x >> 5) + (threadIdx.x >> 5);
  int lane = threadIdx.x & 31;
  int ng  = wave & 7;             // 8 n-groups of 8 n-tiles (NG/16/8)
  int mt  = (wave >> 3) & 2047;   // 2048 M-tiles (NB*NS/16)
  int dir = wave >> 14;
  if (dir > 1) return;

  const __bf16* w   = dir ? w_b   : w_f;
  const float*  bih = dir ? bih_b : bih_f;
  const float*  bhh = dir ? bhh_b : bhh_f;
  __bf16*       out = dir ? xg_b  : xg_f;

  int r0 = mt * 16;
  v8f acc[8];
#pragma unroll
  for (int nn = 0; nn < 8; ++nn) acc[nn] = (v8f){};

#pragma unroll 1
  for (int kt = 0; kt < NE / 32; ++kt) {
    v16bf a = load_tile_bf16(xbf, NE, r0, kt * 32, lane);
#pragma unroll
    for (int nn = 0; nn < 8; ++nn) {
      v16bf b = load_tile_bf16(w, NE, (ng * 8 + nn) * 16, kt * 32, lane);
      acc[nn] = wmma_bf16(a, b, acc[nn]);
    }
  }

  int mh = (lane >> 4) << 3;
#pragma unroll
  for (int nn = 0; nn < 8; ++nn) {
    int n = (ng * 8 + nn) * 16 + (lane & 15);
    float bias = bih[n] + bhh[n];
#pragma unroll
    for (int r = 0; r < 8; ++r) {
      int row = r0 + r + mh;          // = b*NS + s
      int bb  = row >> 9;
      int s   = row & (NS - 1);
      size_t orow = (size_t)s * NB + bb;
      out[orow * NG + n] = (__bf16)(acc[nn][r] + bias);
    }
  }
}

// ---------- kernel 2: persistent LSTM recurrence (one WG per direction) ----------
// LDS: h (bf16 64x256 = 32KB) + gate buffer (bf16 64x1024 = 128KB) -> 160KB/WGP.
// Wave w: mt = w>>3 (fixed) so the LDS h A-tile is reused across its 8 N-tiles.
__global__ void __launch_bounds__(1024, 1)
lstm_seq(const __bf16* __restrict__ whh_f, const __bf16* __restrict__ whh_b,
         const __bf16* __restrict__ xg_f,  const __bf16* __restrict__ xg_b,
         __bf16* __restrict__ lstm_out) {
  __shared__ __bf16 h_lds[NB * NH];
  __shared__ __bf16 g_lds[NB * NG];

  int dir = blockIdx.x;
  const __bf16* whh = dir ? whh_b : whh_f;
  const __bf16* xg  = dir ? xg_b  : xg_f;

  int t    = threadIdx.x;          // 0..1023
  int lane = t & 31;
  int wave = t >> 5;               // 32 waves
  int mt   = wave >> 3;            // 0..3
  int ng   = wave & 7;             // n-group: tiles ng*8 .. ng*8+7

  for (int i = t; i < NB * NH; i += 1024) h_lds[i] = (__bf16)0.f;
  float c_reg[16];
#pragma unroll
  for (int q = 0; q < 16; ++q) c_reg[q] = 0.f;

  int eb = t >> 4;                 // elementwise: batch row
  int eh = (t & 15) * 16;          // elementwise: hidden base (16 each)
  __syncthreads();

  for (int j = 0; j < NS; ++j) {
    int s = dir ? (NS - 1 - j) : j;
    const __bf16* xrow = xg + (size_t)s * NB * NG;

    // ---- phase A: g = h @ whh^T + xg[s]  (M=64, N=1024, K=256) ----
    v8f acc[8];
#pragma unroll
    for (int nn = 0; nn < 8; ++nn) acc[nn] = (v8f){};
    if (j > 0) {
#pragma unroll 1
      for (int kt = 0; kt < NH / 32; ++kt) {
        v16bf a = load_tile_bf16(h_lds, NH, mt * 16, kt * 32, lane);
#pragma unroll
        for (int nn = 0; nn < 8; ++nn) {
          v16bf b = load_tile_bf16(whh, NH, (ng * 8 + nn) * 16, kt * 32, lane);
          acc[nn] = wmma_bf16(a, b, acc[nn]);
        }
      }
    }
    int mh = (lane >> 4) << 3;
#pragma unroll
    for (int nn = 0; nn < 8; ++nn) {
      int n = (ng * 8 + nn) * 16 + (lane & 15);
#pragma unroll
      for (int r = 0; r < 8; ++r) {
        int m = mt * 16 + r + mh;
        float g = acc[nn][r] + (float)xrow[(size_t)m * NG + n];
        g_lds[m * NG + n] = (__bf16)g;
      }
    }
    __syncthreads();

    // ---- phase B: gates -> c,h ; write h to LDS + lstm_out ----
    {
      const __bf16* gi = g_lds + eb * NG;
      __bf16* hrow = h_lds + eb * NH;
      size_t obase = ((size_t)eb * NS + s) * (2 * NH) + dir * NH;
#pragma unroll
      for (int q = 0; q < 16; ++q) {
        int hh  = eh + q;
        float ig = (float)gi[hh];
        float fg = (float)gi[NH + hh];
        float gg = (float)gi[2 * NH + hh];
        float og = (float)gi[3 * NH + hh];
        float c  = sigm(fg) * c_reg[q] + sigm(ig) * tanhf(gg);
        float hn = sigm(og) * tanhf(c);
        c_reg[q] = c;
        __bf16 hb = (__bf16)hn;
        hrow[hh] = hb;
        lstm_out[obase + hh] = hb;
      }
    }
    __syncthreads();
  }
}

// ---------- kernel 3: emissions = lstm_out @ lin_w^T + lin_b (N padded to 16) ----------
__global__ void emissions_gemm(const __bf16* __restrict__ lo,
                               const __bf16* __restrict__ linw,
                               const float* __restrict__ linb,
                               float* __restrict__ em) {
  int wave = blockIdx.x * (blockDim.x >> 5) + (threadIdx.x >> 5);
  int lane = threadIdx.x & 31;
  if (wave >= (NB * NS) / 16) return;
  int r0 = wave * 16;
  v8f acc = {};
#pragma unroll 4
  for (int kt = 0; kt < (2 * NH) / 32; ++kt) {
    v16bf a = load_tile_bf16(lo,   2 * NH, r0, kt * 32, lane);
    v16bf b = load_tile_bf16(linw, 2 * NH, 0,  kt * 32, lane);
    acc = wmma_bf16(a, b, acc);
  }
  int n  = lane & 15;
  int mh = (lane >> 4) << 3;
  float bias = (n < NT) ? linb[n] : 0.f;
#pragma unroll
  for (int r = 0; r < 8; ++r) {
    int row = r0 + r + mh;
    em[(size_t)row * NTP + n] = acc[r] + bias;
  }
}

// ---------- kernel 4: CRF NLL (64 threads, one per batch row) ----------
__global__ void crf_nll(const float* __restrict__ em, const int* __restrict__ target,
                        const int* __restrict__ mask, const float* __restrict__ start_t,
                        const float* __restrict__ end_t, const float* __restrict__ trans,
                        float* __restrict__ out) {
  __shared__ float s_trans[NT * NT], s_start[NT], s_end[NT], s_red[NB];
  int t = threadIdx.x;
  if (t < NT * NT) s_trans[t] = trans[t];
  if (t < NT) { s_start[t] = start_t[t]; s_end[t] = end_t[t]; }
  __syncthreads();

  int b = t;
  const float* eb = em + (size_t)b * NS * NTP;
  const int*   tg = target + (size_t)b * NS;
  const int*   mk = mask + (size_t)b * NS;

  // numerator
  int t0 = tg[0];
  float num = s_start[t0] + eb[t0];
  int msum = mk[0];
  int prev = t0;
  for (int s = 1; s < NS; ++s) {
    int cur = tg[s];
    float mf = (float)mk[s];
    num += (s_trans[prev * NT + cur] + eb[s * NTP + cur]) * mf;
    msum += mk[s];
    prev = cur;
  }
  num += s_end[tg[msum - 1]];

  // partition function (forward algorithm)
  float alpha[NT];
#pragma unroll
  for (int j = 0; j < NT; ++j) alpha[j] = s_start[j] + eb[j];
  for (int s = 1; s < NS; ++s) {
    const float* es = eb + s * NTP;
    float nxt[NT];
#pragma unroll
    for (int j = 0; j < NT; ++j) {
      float mx = -3.4e38f;
#pragma unroll
      for (int i = 0; i < NT; ++i) mx = fmaxf(mx, alpha[i] + s_trans[i * NT + j]);
      float sum = 0.f;
#pragma unroll
      for (int i = 0; i < NT; ++i) sum += __expf(alpha[i] + s_trans[i * NT + j] - mx);
      nxt[j] = mx + __logf(sum) + es[j];
    }
    bool mm = mk[s] != 0;
#pragma unroll
    for (int j = 0; j < NT; ++j) alpha[j] = mm ? nxt[j] : alpha[j];
  }
  float mx = -3.4e38f;
#pragma unroll
  for (int j = 0; j < NT; ++j) mx = fmaxf(mx, alpha[j] + s_end[j]);
  float sum = 0.f;
#pragma unroll
  for (int j = 0; j < NT; ++j) sum += __expf(alpha[j] + s_end[j] - mx);
  s_red[b] = num - (mx + __logf(sum));
  __syncthreads();
  if (t == 0) {
    float acc = 0.f;
    for (int i = 0; i < NB; ++i) acc += s_red[i];
    out[0] = -(acc / (float)NB);
  }
}

// ---------- host launcher ----------
extern "C" void kernel_launch(void* const* d_in, const int* in_sizes, int n_in,
                              void* d_out, int out_size, void* d_ws, size_t ws_size,
                              hipStream_t stream) {
  const float* bert_out = (const float*)d_in[0];
  const int*   mask     = (const int*)d_in[1];
  const int*   target   = (const int*)d_in[2];   // JAX default config -> int32
  const float* w_ih_f   = (const float*)d_in[3];
  const float* w_hh_f   = (const float*)d_in[4];
  const float* b_ih_f   = (const float*)d_in[5];
  const float* b_hh_f   = (const float*)d_in[6];
  const float* w_ih_b   = (const float*)d_in[7];
  const float* w_hh_b   = (const float*)d_in[8];
  const float* b_ih_b   = (const float*)d_in[9];
  const float* b_hh_b   = (const float*)d_in[10];
  const float* lin_w    = (const float*)d_in[11];
  const float* lin_b    = (const float*)d_in[12];
  const float* start_t  = (const float*)d_in[13];
  const float* end_t    = (const float*)d_in[14];
  const float* trans    = (const float*)d_in[15];
  float* out = (float*)d_out;

  char* ws = (char*)d_ws;
  const size_t SZ_X    = (size_t)NB * NS * NE * 2;
  const size_t SZ_WIH  = (size_t)NG * NE * 2;
  const size_t SZ_WHH  = (size_t)NG * NH * 2;
  const size_t SZ_LINW = (size_t)NTP * 2 * NH * 2;
  const size_t SZ_XG   = (size_t)NS * NB * NG * 2;
  const size_t SZ_LO   = (size_t)NB * NS * 2 * NH * 2;
  size_t o = 0;
  __bf16* x_bf    = (__bf16*)(ws + o); o += SZ_X;
  __bf16* wihf_bf = (__bf16*)(ws + o); o += SZ_WIH;
  __bf16* wihb_bf = (__bf16*)(ws + o); o += SZ_WIH;
  __bf16* whhf_bf = (__bf16*)(ws + o); o += SZ_WHH;
  __bf16* whhb_bf = (__bf16*)(ws + o); o += SZ_WHH;
  __bf16* linw_bf = (__bf16*)(ws + o); o += SZ_LINW;
  __bf16* xg_f    = (__bf16*)(ws + o); o += SZ_XG;
  __bf16* xg_b    = (__bf16*)(ws + o); o += SZ_XG;
  __bf16* lo      = (__bf16*)(ws + o); o += SZ_LO;
  float*  em      = (float*)(ws + o);

  // 0) bf16 conversions (activations once; weights once)
  {
    int n = NB * NS * NE;
    cvt_f32_to_bf16<<<(n + 255) / 256, 256, 0, stream>>>(bert_out, x_bf, n);
    n = NG * NE;
    cvt_f32_to_bf16<<<(n + 255) / 256, 256, 0, stream>>>(w_ih_f, wihf_bf, n);
    cvt_f32_to_bf16<<<(n + 255) / 256, 256, 0, stream>>>(w_ih_b, wihb_bf, n);
    n = NG * NH;
    cvt_f32_to_bf16<<<(n + 255) / 256, 256, 0, stream>>>(w_hh_f, whhf_bf, n);
    cvt_f32_to_bf16<<<(n + 255) / 256, 256, 0, stream>>>(w_hh_b, whhb_bf, n);
    pad_linw<<<(NTP * 2 * NH + 255) / 256, 256, 0, stream>>>(lin_w, linw_bf);
  }
  // 1) big input-gate GEMM, both directions: 2 * 2048 * 8 waves, 8 waves/block
  xg_gemm<<<(2 * 2048 * 8) / 8, 256, 0, stream>>>(
      x_bf, wihf_bf, wihb_bf, b_ih_f, b_hh_f, b_ih_b, b_hh_b, xg_f, xg_b);
  // 2) sequential recurrence, one persistent workgroup per direction
  lstm_seq<<<2, 1024, 0, stream>>>(whhf_bf, whhb_bf, xg_f, xg_b, lo);
  // 3) emissions GEMM (2048 waves / 8 per block)
  emissions_gemm<<<2048 / 8, 256, 0, stream>>>(lo, linw_bf, lin_b, em);
  // 4) CRF scalar NLL
  crf_nll<<<1, NB, 0, stream>>>(em, target, mask, start_t, end_t, trans, out);
}
